// CosineAttention_68865505624685
// MI455X (gfx1250) — matrix operand order repaired and checked by reading, not compiled, not run on hardware
//
#include <hip/hip_runtime.h>

// ---------------------------------------------------------------------------
// Cosine attention (flash-attention style) for MI455X / gfx1250.
//   q,k,v: [N,L|S,H,D] f32.  out: [N,L,H,D] f32.
//   Compute-bound (AI ~500 FLOP/B) -> bf16 WMMA (v_wmma_f32_16x16x32_bf16)
//   with fp32 accumulation; K/V staged via global_load_async_to_lds_b128
//   (ASYNCcnt) with double buffering. Softmax row reductions use DPP16
//   butterflies (quad_perm / row_half_mirror / row_mirror) instead of
//   ds_bpermute so the LDS pipe stays free for WMMA fragment loads.
// ---------------------------------------------------------------------------

typedef __attribute__((ext_vector_type(16))) __bf16 v16bf;
typedef __attribute__((ext_vector_type(8)))  __bf16 v8bf;
typedef __attribute__((ext_vector_type(8)))  float  v8f;

constexpr int Nb = 4, Lq = 2048, Sk = 2048, Hh = 8, Dd = 64;
constexpr int BLOCK_L = 128;   // 8 waves * 16 rows
constexpr int STILE   = 64;    // keys per S-tile
constexpr int THREADS = 256;
constexpr int NWAVE   = 8;

__device__ __forceinline__ unsigned lds_off(const void* p) {
  // low 32 bits of a flat shared-aperture address == LDS byte offset
  return (unsigned)(unsigned long long)p;
}

__device__ __forceinline__ void async_b128(unsigned lds, unsigned long long gaddr) {
  asm volatile("global_load_async_to_lds_b128 %0, %1, off"
               :: "v"(lds), "v"(gaddr) : "memory");
}

__device__ __forceinline__ void wait_async0() {
  asm volatile("s_wait_asynccnt 0" ::: "memory");
}

// ---- DPP16 lane exchanges (VALU only, no LDS) -----------------------------
// ctrl: 0x00-0xFF quad_perm, 0x140 row_mirror (xor15 in 16),
//       0x141 row_half_mirror (xor7 in 8)
template <int CTRL>
__device__ __forceinline__ float dpp_movf(float x) {
#if __has_builtin(__builtin_amdgcn_mov_dpp)
  int r = __builtin_amdgcn_mov_dpp(__builtin_bit_cast(int, x),
                                   CTRL, 0xF, 0xF, true);
  return __builtin_bit_cast(float, r);
#else
  return __shfl_xor(x, (CTRL == 0xB1) ? 1 : (CTRL == 0x4E) ? 2
                                         : (CTRL == 0x141) ? 4 : 8, 32);
#endif
}

// butterfly reduce over the 16 lanes of a half-wave
__device__ __forceinline__ float red_max16(float x) {
  x = fmaxf(x, dpp_movf<0xB1>(x));   // quad_perm(1,0,3,2)  == xor1
  x = fmaxf(x, dpp_movf<0x4E>(x));   // quad_perm(2,3,0,1)  == xor2
  x = fmaxf(x, dpp_movf<0x141>(x));  // row_half_mirror: xor7 ≡ xor4 now
  x = fmaxf(x, dpp_movf<0x140>(x));  // row_mirror:      xor15 ≡ xor8 now
  return x;
}
__device__ __forceinline__ float red_add16(float x) {
  x += dpp_movf<0xB1>(x);
  x += dpp_movf<0x4E>(x);
  x += dpp_movf<0x141>(x);
  x += dpp_movf<0x140>(x);
  return x;
}

// swap the two 16-lane halves of the wave (xor16)
__device__ __forceinline__ float xhalf_swap(float x) {
#if __has_builtin(__builtin_amdgcn_permlanex16)
  int xi = __builtin_bit_cast(int, x);
  int r = __builtin_amdgcn_permlanex16(xi, xi, 0x76543210, 0xFEDCBA98,
                                       false, false);
  return __builtin_bit_cast(float, r);
#else
  return __shfl_xor(x, 16, 32);
#endif
}

__global__ __launch_bounds__(THREADS)
void cosattn_wmma_kernel(const float* __restrict__ q,
                         const float* __restrict__ k,
                         const float* __restrict__ v,
                         const float* __restrict__ temp,
                         float* __restrict__ out)
{
  // raw fp32 staging (async target), double buffered
  __shared__ __align__(16) float  stK[2][STILE * Dd];   // 2 x 16 KB
  __shared__ __align__(16) float  stV[2][STILE * Dd];   // 2 x 16 KB
  // bf16 compute tiles
  __shared__ __align__(32) __bf16 Kbf[STILE * Dd];      // [s][d]   8 KB
  __shared__ __align__(32) __bf16 Vt [Dd * STILE];      // [d][s]   8 KB
  __shared__ __align__(32) __bf16 Pb [NWAVE][16 * STILE]; // per-wave P, 16 KB

  const int t    = threadIdx.x;
  const int lane = t & 31;
  const int wv   = t >> 5;
  const int nlo  = lane & 15;
  const int hi   = lane >> 4;

  const int LT = Lq / BLOCK_L;              // 16
  const int bx = blockIdx.x;
  const int lt = bx % LT;
  const int h  = (bx / LT) % Hh;
  const int n  = bx / (LT * Hh);
  const int l0 = lt * BLOCK_L;

  const float invT = 1.0f / temp[0];

  // ---- load + normalize Q rows into A-matrix bf16 fragments -------------
  // A 16x32 (16-bit) per-lane layout used here: 16 contiguous K values
  // [16*hi..16*hi+15] within each 32-wide K chunk; row M = lane&15.
  const int gl = l0 + wv * 16 + nlo;
  const float* qp = q + (((size_t)n * Lq + gl) * Hh + h) * Dd;
  float qv[4][8];
  float ss = 0.f;
  #pragma unroll
  for (int r = 0; r < 4; ++r) {
    const int db = (r >> 1) * 32 + 16 * hi + (r & 1) * 8; // 16hi, 16hi+8, 32+16hi, 40+16hi
    float4 a = *(const float4*)(qp + db);
    float4 b = *(const float4*)(qp + db + 4);
    qv[r][0]=a.x; qv[r][1]=a.y; qv[r][2]=a.z; qv[r][3]=a.w;
    qv[r][4]=b.x; qv[r][5]=b.y; qv[r][6]=b.z; qv[r][7]=b.w;
    #pragma unroll
    for (int i = 0; i < 8; ++i) ss += qv[r][i] * qv[r][i];
  }
  ss += xhalf_swap(ss);                     // combine the two half-rows
  const float rq = rsqrtf(ss);
  union { v16bf w; __bf16 e[16]; } uq0, uq1;
  #pragma unroll
  for (int i = 0; i < 8; ++i) {
    uq0.e[i]     = (__bf16)(qv[0][i] * rq);
    uq0.e[8 + i] = (__bf16)(qv[1][i] * rq);
    uq1.e[i]     = (__bf16)(qv[2][i] * rq);
    uq1.e[8 + i] = (__bf16)(qv[3][i] * rq);
  }
  const v16bf qa0 = uq0.w, qa1 = uq1.w;     // d 0..31, d 32..63

  // ---- online-softmax state: rows m = vr + 8*hi --------------------------
  float rmax[8], rsum[8];
  #pragma unroll
  for (int i = 0; i < 8; ++i) { rmax[i] = -1e30f; rsum[i] = 0.f; }
  v8f acc[4] = {};                          // out accumulators, 4 d-tiles

  const float* kbase = k + (((size_t)n * Sk) * Hh + h) * Dd;
  const float* vbase = v + (((size_t)n * Sk) * Hh + h) * Dd;
  const int rowStride = Hh * Dd;            // floats between consecutive s

  auto prefetch = [&](int st, int buf) {
    const float* kb = kbase + (size_t)st * STILE * rowStride;
    const float* vb = vbase + (size_t)st * STILE * rowStride;
    #pragma unroll
    for (int i = 0; i < 4; ++i) {
      const int idx = t + i * THREADS;      // 0..1023 transfers per matrix
      const int row = idx >> 4;             // key row 0..63
      const int seg = idx & 15;             // 16B segment within row
      async_b128(lds_off(&stK[buf][row * Dd + seg * 4]),
                 (unsigned long long)(kb + row * rowStride + seg * 4));
      async_b128(lds_off(&stV[buf][row * Dd + seg * 4]),
                 (unsigned long long)(vb + row * rowStride + seg * 4));
    }
  };

  prefetch(0, 0);
  constexpr int NST = Sk / STILE;           // 32 S-tiles

  for (int st = 0; st < NST; ++st) {
    const int buf = st & 1;
    wait_async0();
    __syncthreads();                        // staging ready; prev compute done

    // ---- convert: normalize K -> Kbf[s][d]; transpose V -> Vt[d][s] ----
    {
      const int kr = t >> 2;                // key row 0..63
      const int kq = t & 3;                 // 16-element quarter
      const float* srow = &stK[buf][kr * Dd + kq * 16];
      float kv[16]; float ks = 0.f;
      #pragma unroll
      for (int i = 0; i < 16; ++i) { kv[i] = srow[i]; ks += kv[i] * kv[i]; }
      ks += dpp_movf<0xB1>(ks);             // xor1
      ks += dpp_movf<0x4E>(ks);             // xor2 -> full row sumsq
      const float rk = rsqrtf(ks);
      union { v16bf w; __bf16 e[16]; } ub;
      #pragma unroll
      for (int i = 0; i < 16; ++i) ub.e[i] = (__bf16)(kv[i] * rk);
      *(v16bf*)&Kbf[kr * Dd + kq * 16] = ub.w;

      const float* vrow = &stV[buf][kr * Dd + kq * 16];
      #pragma unroll
      for (int i = 0; i < 16; ++i)
        Vt[(kq * 16 + i) * STILE + kr] = (__bf16)vrow[i];
    }
    __syncthreads();                        // Kbf/Vt ready

    if (st + 1 < NST) prefetch(st + 1, buf ^ 1);  // overlap next tile's DMA

    // ---- scores: S_tile = Qhat(16x64) * Khat^T(64x64), 4 col-tiles -----
    // B 32x16 (16-bit) per-lane: 16 contiguous K values of column n=lane&15,
    // K base = 16*hi.
    v8f sc[4];
    #pragma unroll
    for (int nt = 0; nt < 4; ++nt) {
      const int col = nt * 16 + nlo;        // key index within tile
      const v16bf b0 = *(const v16bf*)&Kbf[col * Dd + 16 * hi];
      const v16bf b1 = *(const v16bf*)&Kbf[col * Dd + 32 + 16 * hi];
      v8f c = {};
      c = __builtin_amdgcn_wmma_f32_16x16x32_bf16(false, qa0, false, b0,
                                                  (short)0, c, false, false);
      c = __builtin_amdgcn_wmma_f32_16x16x32_bf16(false, qa1, false, b1,
                                                  (short)0, c, false, false);
      sc[nt] = c;
    }

    // ---- online softmax (C layout: row = vr + 8*hi, col = nt*16+nlo) ---
    #pragma unroll
    for (int vr = 0; vr < 8; ++vr) {
      float s0 = sc[0][vr] * invT, s1 = sc[1][vr] * invT;
      float s2 = sc[2][vr] * invT, s3 = sc[3][vr] * invT;
      float tm = fmaxf(fmaxf(s0, s1), fmaxf(s2, s3));
      tm = red_max16(tm);                   // row max across 16 columns
      const float nm   = fmaxf(rmax[vr], tm);
      const float fold = __expf(rmax[vr] - nm);
      rmax[vr] = nm;
      const float p0 = __expf(s0 - nm), p1 = __expf(s1 - nm);
      const float p2 = __expf(s2 - nm), p3 = __expf(s3 - nm);
      float psum = red_add16(p0 + p1 + p2 + p3);
      rsum[vr] = rsum[vr] * fold + psum;
      #pragma unroll
      for (int d = 0; d < 4; ++d) acc[d][vr] *= fold;
      __bf16* pr = &Pb[wv][(vr + 8 * hi) * STILE + nlo];
      pr[0]  = (__bf16)p0;
      pr[16] = (__bf16)p1;
      pr[32] = (__bf16)p2;
      pr[48] = (__bf16)p3;
    }

    // ---- acc += P(16x64) * V(64x64): 2 K-chunks x 4 d-tiles ------------
    #pragma unroll
    for (int kc = 0; kc < 2; ++kc) {
      union { v16bf w; v8bf hh[2]; } up;     // A layout: two 8-elem runs
      up.hh[0] = *(const v8bf*)&Pb[wv][nlo * STILE + kc * 32 + 16 * hi];
      up.hh[1] = *(const v8bf*)&Pb[wv][nlo * STILE + kc * 32 + 16 * hi + 8];
      const v16bf pa = up.w;
      #pragma unroll
      for (int dt = 0; dt < 4; ++dt) {
        const v16bf vb2 =
            *(const v16bf*)&Vt[(dt * 16 + nlo) * STILE + kc * 32 + 16 * hi];
        acc[dt] = __builtin_amdgcn_wmma_f32_16x16x32_bf16(
            false, pa, false, vb2, (short)0, acc[dt], false, false);
      }
    }
  }

  // ---- epilogue: divide by softmax denominator, store --------------------
  #pragma unroll
  for (int vr = 0; vr < 8; ++vr) {
    const int row  = vr + 8 * hi;
    const float iv = 1.0f / rsum[vr];
    float* op = out + (((size_t)n * Lq + (l0 + wv * 16 + row)) * Hh + h) * Dd;
    #pragma unroll
    for (int dt = 0; dt < 4; ++dt) op[dt * 16 + nlo] = acc[dt][vr] * iv;
  }
}

extern "C" void kernel_launch(void* const* d_in, const int* in_sizes, int n_in,
                              void* d_out, int out_size, void* d_ws, size_t ws_size,
                              hipStream_t stream) {
  (void)in_sizes; (void)n_in; (void)out_size; (void)d_ws; (void)ws_size;
  const float* q    = (const float*)d_in[0];
  const float* k    = (const float*)d_in[1];
  const float* v    = (const float*)d_in[2];
  const float* temp = (const float*)d_in[3];
  float* out = (float*)d_out;

  dim3 grid(Nb * Hh * (Lq / BLOCK_L));   // 512 blocks
  dim3 block(THREADS);                   // 8 waves (wave32)
  hipLaunchKernelGGL(cosattn_wmma_kernel, grid, block, 0, stream,
                     q, k, v, temp, out);
}
